// LinguistBiLSTM_CRF_39341900432003
// MI455X (gfx1250) — matrix operand order repaired and checked by reading, not compiled, not run on hardware
//
#include <hip/hip_runtime.h>
#include <stddef.h>

// ---------------------------------------------------------------------------
// Types for CDNA5 WMMA
// ---------------------------------------------------------------------------
typedef __attribute__((ext_vector_type(16))) __bf16 v16bf;
typedef __attribute__((ext_vector_type(8)))  float  v8f;

union BF16x16 { v16bf v; uint4 q[2]; };

// Problem constants
#define VSZ 50000
#define TT  32
#define EE  256
#define HH  512
#define G4  2048   // 4*H
#define BB  64
#define SS  512

__device__ __forceinline__ unsigned short f2bf(float f) {
    union { float f; unsigned int u; } x; x.f = f;
    unsigned int u = x.u;
    u += 0x7fffu + ((u >> 16) & 1u);   // round-to-nearest-even
    return (unsigned short)(u >> 16);
}

// ---------------------------------------------------------------------------
// WMMA fragment loads.  Both operands stored K-contiguous (row-major, ld=K).
// A: 16x32 (MxK) bf16.  lane[3:0]=M row; lane[4] selects K+8 half;
//    elements 0..7 -> K = kb..kb+7, elements 8..15 -> K = kb+16..kb+23.
// B: 32x16 (KxN) bf16, sourced from W stored as N x K (so B[k][n]=W[n][k]).
//    lane[3:0]=N col; lane[4] selects K+8 half; same K order as A.
// ---------------------------------------------------------------------------
__device__ __forceinline__ v16bf load_a_frag(const unsigned short* A, int lda,
                                             int m0, int k0, int lane) {
    int m  = m0 + (lane & 15);
    int kb = k0 + ((lane & 16) ? 8 : 0);
    const unsigned short* p = A + (size_t)m * lda + kb;
    BF16x16 u;
    u.q[0] = *reinterpret_cast<const uint4*>(p);        // K = kb .. kb+7
    u.q[1] = *reinterpret_cast<const uint4*>(p + 16);   // K = kb+16 .. kb+23
    return u.v;
}

__device__ __forceinline__ v16bf load_b_frag(const unsigned short* W, int ldw,
                                             int n0, int k0, int lane) {
    int n  = n0 + (lane & 15);
    int kb = k0 + ((lane & 16) ? 8 : 0);
    const unsigned short* p = W + (size_t)n * ldw + kb;
    BF16x16 u;
    u.q[0] = *reinterpret_cast<const uint4*>(p);        // K = kb .. kb+7
    u.q[1] = *reinterpret_cast<const uint4*>(p + 16);   // K = kb+16 .. kb+23
    return u.v;
}

// ---------------------------------------------------------------------------
// f32 -> bf16 conversion
// ---------------------------------------------------------------------------
__global__ void k_f32_to_bf16(const float* __restrict__ src,
                              unsigned short* __restrict__ dst, int n) {
    int i = blockIdx.x * blockDim.x + threadIdx.x;
    if (i < n) dst[i] = f2bf(src[i]);
}

// ---------------------------------------------------------------------------
// Embedding gather:  e_bf[(s*B+b)*E + k] = bf16(emb[x[b][s]][k])
// ---------------------------------------------------------------------------
__global__ void k_gather_emb(const float* __restrict__ emb,
                             const int* __restrict__ x,
                             unsigned short* __restrict__ e_bf) {
    size_t idx = (size_t)blockIdx.x * blockDim.x + threadIdx.x; // < S*B*E
    int k   = (int)(idx & (EE - 1));
    int row = (int)(idx >> 8);      // s*B + b
    int b   = row & (BB - 1);
    int s   = row >> 6;
    int tok = x[b * SS + s];
    e_bf[idx] = f2bf(emb[(size_t)tok * EE + k]);
}

// ---------------------------------------------------------------------------
// Zero-init h (bf16) and c (f32) states: 2*64*512 elements each
// ---------------------------------------------------------------------------
__global__ void k_init_state(float* __restrict__ c,
                             unsigned short* __restrict__ h_bf) {
    int i = blockIdx.x * blockDim.x + threadIdx.x;  // < 65536
    c[i] = 0.f;
    h_bf[i] = 0;
}

// ---------------------------------------------------------------------------
// LSTM step GEMM (both directions):
//   gates[dir] = [e_t , h_prev] @ [W_ih , W_hh]^T + bias      (64 x 2048)
// 1024 waves: dir(2) x mtile(4) x ntile(128); 24 WMMA per wave (K=768).
// ---------------------------------------------------------------------------
__global__ void __launch_bounds__(256)
k_lstm_step_gemm(const unsigned short* __restrict__ e_bf,
                 const unsigned short* __restrict__ h_bf,
                 const unsigned short* __restrict__ wih_bf,
                 const unsigned short* __restrict__ whh_bf,
                 const float* __restrict__ bias_f,
                 const float* __restrict__ bias_b,
                 float* __restrict__ gates, int s) {
    int lane  = threadIdx.x;
    int wid   = blockIdx.x * blockDim.y + threadIdx.y;  // 0..1023
    int dir   = wid >> 9;
    int tile  = wid & 511;
    int m0    = (tile & 3) * 16;     // batch-row tile (M=64)
    int n0    = (tile >> 2) * 16;    // gate-col tile  (N=2048)
    int t     = dir ? (SS - 1 - s) : s;

    const unsigned short* eA  = e_bf + (size_t)t * BB * EE;          // 64 x 256
    const unsigned short* hA  = h_bf + (size_t)dir * BB * HH;        // 64 x 512
    const unsigned short* Wih = wih_bf + (size_t)dir * G4 * EE;      // 2048 x 256
    const unsigned short* Whh = whh_bf + (size_t)dir * G4 * HH;      // 2048 x 512

    v8f acc = {};
    for (int k = 0; k < EE; k += 32) {
        v16bf a = load_a_frag(eA, EE, m0, k, lane);
        v16bf b = load_b_frag(Wih, EE, n0, k, lane);
        acc = __builtin_amdgcn_wmma_f32_16x16x32_bf16(false, a, false, b,
                                                      (short)0, acc, false, false);
    }
    for (int k = 0; k < HH; k += 32) {
        v16bf a = load_a_frag(hA, HH, m0, k, lane);
        v16bf b = load_b_frag(Whh, HH, n0, k, lane);
        acc = __builtin_amdgcn_wmma_f32_16x16x32_bf16(false, a, false, b,
                                                      (short)0, acc, false, false);
    }

    const float* bias = dir ? bias_b : bias_f;
    int n  = n0 + (lane & 15);
    float bv = bias[n];
    float* g = gates + (size_t)dir * BB * G4;
    int mb = m0 + ((lane & 16) ? 8 : 0);
#pragma unroll
    for (int r = 0; r < 8; ++r)
        g[(size_t)(mb + r) * G4 + n] = acc[r] + bv;
}

// ---------------------------------------------------------------------------
// LSTM gate update: c,h update for both directions. 65536 threads.
// Gate order from jnp.split: i | f | g | o
// ---------------------------------------------------------------------------
__global__ void k_lstm_update(const float* __restrict__ gates,
                              float* __restrict__ c,
                              unsigned short* __restrict__ h_bf,
                              unsigned short* __restrict__ h_all, int s) {
    int idx = blockIdx.x * blockDim.x + threadIdx.x;   // < 2*64*512
    int dir = idx >> 15;
    int rem = idx & 32767;
    int b   = rem >> 9;
    int j   = rem & 511;

    const float* g = gates + ((size_t)dir * BB + b) * G4;
    float gi = g[j], gf = g[HH + j], gg = g[2 * HH + j], go = g[3 * HH + j];
    float i_ = 1.f / (1.f + __expf(-gi));
    float f_ = 1.f / (1.f + __expf(-gf));
    float o_ = 1.f / (1.f + __expf(-go));
    float cn = f_ * c[idx] + i_ * tanhf(gg);
    float h  = o_ * tanhf(cn);
    c[idx]   = cn;
    h_bf[idx] = f2bf(h);
    int t = dir ? (SS - 1 - s) : s;
    h_all[((size_t)t * BB + b) * (2 * HH) + dir * HH + j] = f2bf(h);
}

// ---------------------------------------------------------------------------
// FC GEMM: em = h_all(32768 x 1024) @ fc_w^T(1024 x 32) + fc_b  -> f32
// 4096 waves: mtile(2048) x ntile(2); 32 WMMA per wave.
// ---------------------------------------------------------------------------
__global__ void __launch_bounds__(256)
k_fc_gemm(const unsigned short* __restrict__ h_all,
          const unsigned short* __restrict__ fcw_bf,
          const float* __restrict__ fc_b,
          float* __restrict__ em) {
    int lane = threadIdx.x;
    int wid  = blockIdx.x * blockDim.y + threadIdx.y;  // 0..4095
    int n0   = (wid & 1) * 16;
    int m0   = (wid >> 1) * 16;

    v8f acc = {};
    for (int k = 0; k < 2 * HH; k += 32) {
        v16bf a = load_a_frag(h_all, 2 * HH, m0, k, lane);
        v16bf b = load_b_frag(fcw_bf, 2 * HH, n0, k, lane);
        acc = __builtin_amdgcn_wmma_f32_16x16x32_bf16(false, a, false, b,
                                                      (short)0, acc, false, false);
    }
    int n  = n0 + (lane & 15);
    float bv = fc_b[n];
    int mb = m0 + ((lane & 16) ? 8 : 0);
#pragma unroll
    for (int r = 0; r < 8; ++r)
        em[(size_t)(mb + r) * TT + n] = acc[r] + bv;
}

// ---------------------------------------------------------------------------
// CRF numerator: one block per batch element
// ---------------------------------------------------------------------------
__global__ void k_crf_num(const float* __restrict__ em,
                          const float* __restrict__ trans,
                          const float* __restrict__ start_t,
                          const float* __restrict__ end_t,
                          const int* __restrict__ tags,
                          float* __restrict__ num) {
    int b = blockIdx.x;
    __shared__ float red[256];
    float acc = 0.f;
    for (int s = threadIdx.x; s < SS; s += 256) {
        int tg = tags[b * SS + s];
        acc += em[((size_t)s * BB + b) * TT + tg];
        if (s < SS - 1) acc += trans[tg * TT + tags[b * SS + s + 1]];
    }
    red[threadIdx.x] = acc;
    __syncthreads();
    for (int st = 128; st > 0; st >>= 1) {
        if (threadIdx.x < st) red[threadIdx.x] += red[threadIdx.x + st];
        __syncthreads();
    }
    if (threadIdx.x == 0)
        num[b] = red[0] + start_t[tags[b * SS]] + end_t[tags[b * SS + SS - 1]];
}

// ---------------------------------------------------------------------------
// CRF forward algorithm: one wave32 per batch element, lane = tag.
// ---------------------------------------------------------------------------
__global__ void k_crf_forward(const float* __restrict__ em,
                              const float* __restrict__ trans,
                              const float* __restrict__ start_t,
                              const float* __restrict__ end_t,
                              const float* __restrict__ num,
                              float* __restrict__ llh) {
    int lane = threadIdx.x;                 // tag t'
    int b    = blockIdx.x * blockDim.y + threadIdx.y;   // batch

    float trcol[TT];
#pragma unroll
    for (int t = 0; t < TT; ++t) trcol[t] = trans[t * TT + lane];

    float alpha = start_t[lane] + em[(size_t)b * TT + lane];

    for (int s = 1; s < SS; ++s) {
        float et = em[((size_t)s * BB + b) * TT + lane];
        float v[TT];
        float m = -3.4e38f;
#pragma unroll
        for (int t = 0; t < TT; ++t) {
            v[t] = __shfl(alpha, t, 32) + trcol[t];
            m = fmaxf(m, v[t]);
        }
        float ssum = 0.f;
#pragma unroll
        for (int t = 0; t < TT; ++t) ssum += __expf(v[t] - m);
        alpha = m + __logf(ssum) + et;
    }

    float val = alpha + end_t[lane];
    float m = val;
#pragma unroll
    for (int off = 16; off > 0; off >>= 1) m = fmaxf(m, __shfl_xor(m, off, 32));
    float e = __expf(val - m);
#pragma unroll
    for (int off = 16; off > 0; off >>= 1) e += __shfl_xor(e, off, 32);
    float den = m + __logf(e);
    if (lane == 0) llh[b] = num[b] - den;
}

// ---------------------------------------------------------------------------
// Final: out = -mean(llh)
// ---------------------------------------------------------------------------
__global__ void k_final(const float* __restrict__ llh, float* __restrict__ out) {
    __shared__ float red[BB];
    red[threadIdx.x] = llh[threadIdx.x];
    __syncthreads();
    for (int st = 32; st > 0; st >>= 1) {
        if (threadIdx.x < st) red[threadIdx.x] += red[threadIdx.x + st];
        __syncthreads();
    }
    if (threadIdx.x == 0) out[0] = -red[0] / (float)BB;
}

// ---------------------------------------------------------------------------
// Host orchestration
// ---------------------------------------------------------------------------
extern "C" void kernel_launch(void* const* d_in, const int* in_sizes, int n_in,
                              void* d_out, int out_size, void* d_ws, size_t ws_size,
                              hipStream_t stream) {
    (void)in_sizes; (void)n_in; (void)out_size; (void)ws_size;
    const float* emb     = (const float*)d_in[0];
    const float* w_ih_f  = (const float*)d_in[1];
    const float* w_hh_f  = (const float*)d_in[2];
    const float* b_f     = (const float*)d_in[3];
    const float* w_ih_b  = (const float*)d_in[4];
    const float* w_hh_b  = (const float*)d_in[5];
    const float* b_b     = (const float*)d_in[6];
    const float* fc_w    = (const float*)d_in[7];
    const float* fc_b    = (const float*)d_in[8];
    const float* start_t = (const float*)d_in[9];
    const float* end_t   = (const float*)d_in[10];
    const float* trans   = (const float*)d_in[11];
    const int*   x       = (const int*)d_in[12];
    const int*   tags    = (const int*)d_in[13];
    float* out = (float*)d_out;

    char* ws = (char*)d_ws;
    size_t off = 0;
    auto carve = [&](size_t bytes) -> char* {
        char* p = ws + off;
        off = (off + bytes + 255) & ~(size_t)255;
        return p;
    };
    unsigned short* e_bf   = (unsigned short*)carve((size_t)SS * BB * EE * 2);  // 16 MB
    unsigned short* wih_bf = (unsigned short*)carve((size_t)2 * G4 * EE * 2);   //  2 MB
    unsigned short* whh_bf = (unsigned short*)carve((size_t)2 * G4 * HH * 2);   //  4 MB
    unsigned short* fcw_bf = (unsigned short*)carve((size_t)TT * 2 * HH * 2);   // 64 KB
    float*          gates  = (float*)carve((size_t)2 * BB * G4 * 4);            //  1 MB
    unsigned short* h_bf   = (unsigned short*)carve((size_t)2 * BB * HH * 2);   // 128 KB
    float*          c_st   = (float*)carve((size_t)2 * BB * HH * 4);            // 256 KB
    unsigned short* h_all  = (unsigned short*)carve((size_t)SS * BB * 2 * HH * 2); // 64 MB
    float*          em     = (float*)carve((size_t)SS * BB * TT * 4);           //  4 MB
    float*          num    = (float*)carve((size_t)BB * 4);
    float*          llh    = (float*)carve((size_t)BB * 4);

    // Weight conversions to bf16 (resident in L2 thereafter)
    k_f32_to_bf16<<<dim3(2048), dim3(256), 0, stream>>>(w_ih_f, wih_bf,               G4 * EE);
    k_f32_to_bf16<<<dim3(2048), dim3(256), 0, stream>>>(w_ih_b, wih_bf + G4 * EE,     G4 * EE);
    k_f32_to_bf16<<<dim3(4096), dim3(256), 0, stream>>>(w_hh_f, whh_bf,               G4 * HH);
    k_f32_to_bf16<<<dim3(4096), dim3(256), 0, stream>>>(w_hh_b, whh_bf + G4 * HH,     G4 * HH);
    k_f32_to_bf16<<<dim3(128),  dim3(256), 0, stream>>>(fc_w,   fcw_bf,               TT * 2 * HH);

    // Embedding gather (S*B*E = 8.4M elements)
    k_gather_emb<<<dim3(32768), dim3(256), 0, stream>>>(emb, x, e_bf);

    // Zero LSTM states
    k_init_state<<<dim3(256), dim3(256), 0, stream>>>(c_st, h_bf);

    // Bidirectional LSTM scan (fused input projection, WMMA bf16)
    for (int s = 0; s < SS; ++s) {
        k_lstm_step_gemm<<<dim3(128), dim3(32, 8), 0, stream>>>(
            e_bf, h_bf, wih_bf, whh_bf, b_f, b_b, gates, s);
        k_lstm_update<<<dim3(256), dim3(256), 0, stream>>>(gates, c_st, h_bf, h_all, s);
    }

    // Emission scores
    k_fc_gemm<<<dim3(512), dim3(32, 8), 0, stream>>>(h_all, fcw_bf, fc_b, em);

    // CRF
    k_crf_num<<<dim3(BB), dim3(256), 0, stream>>>(em, trans, start_t, end_t, tags, num);
    k_crf_forward<<<dim3(16), dim3(32, 4), 0, stream>>>(em, trans, start_t, end_t, num, llh);
    k_final<<<dim3(1), dim3(BB), 0, stream>>>(llh, out);
}
